// PointNetFeaturePropagation_23270132810093
// MI455X (gfx1250) — matrix operand order repaired and checked by reading, not compiled, not run on hardware
//
#include <hip/hip_runtime.h>

// ---------------- problem constants ----------------
#define B_    4
#define N_    16384
#define S_    4096
#define C1_   128
#define C2_   256
#define CIN_  384     // C1 + C2
#define H1_   256     // mlp layer 1 out
#define COUT_ 128     // mlp layer 2 out
#define R_    (B_*N_) // 65536 total points

typedef __attribute__((ext_vector_type(16))) __bf16 v16bf;
typedef __attribute__((ext_vector_type(8)))  float  v8f;

// ---------------- helpers ----------------
__device__ __forceinline__ unsigned short f2bf(float x) {
  unsigned u = __float_as_uint(x);
  return (unsigned short)((u + 0x7FFFu + ((u >> 16) & 1u)) >> 16); // RNE
}
__device__ __forceinline__ unsigned pack2bf(float a, float b) {
  return (unsigned)f2bf(a) | ((unsigned)f2bf(b) << 16);
}

// gfx1250 async global->LDS copy of a 64B row (4 x b128), ASYNCcnt-tracked.
// INST_OFFSET is added to BOTH the LDS dest and the global src address, so a
// single base pair + immediate offsets covers the whole row.
__device__ __forceinline__ void async_g2l_row64(unsigned lds_byte_addr, const void* gaddr) {
  asm volatile(
      "global_load_async_to_lds_b128 %0, %1, off\n\t"
      "global_load_async_to_lds_b128 %0, %1, off offset:16\n\t"
      "global_load_async_to_lds_b128 %0, %1, off offset:32\n\t"
      "global_load_async_to_lds_b128 %0, %1, off offset:48"
      :: "v"(lds_byte_addr), "v"(gaddr) : "memory");
}
__device__ __forceinline__ void wait_async0() {
  asm volatile("s_wait_asynccnt 0x0" ::: "memory");
}
__device__ __forceinline__ unsigned lds_addr_of(const void* p) {
  // generic pointer to LDS: low 32 bits are the LDS byte address
  return (unsigned)(unsigned long long)p;
}

// ================= kernel 1: weight f32 -> bf16 =================
__global__ void __launch_bounds__(256) cvt_weights(
    const float* __restrict__ W1, const float* __restrict__ W2,
    unsigned* __restrict__ W1b, unsigned* __restrict__ W2b) {
  int i = blockIdx.x * blockDim.x + threadIdx.x; // pair index
  const int n1 = (H1_ * CIN_) / 2;   // 49152
  const int n2 = (COUT_ * H1_) / 2;  // 16384
  if (i < n1) W1b[i] = pack2bf(W1[2*i], W1[2*i+1]);
  if (i < n2) W2b[i] = pack2bf(W2[2*i], W2[2*i+1]);
}

// ================= kernel 2: brute-force 3-NN =================
#define KTS 512
__global__ void __launch_bounds__(256) knn3(
    const float* __restrict__ xyz1, const float* __restrict__ xyz2,
    int* __restrict__ idx, float* __restrict__ wgt) {
  __shared__ float sx[KTS], sy[KTS], sz[KTS];
  const int b = blockIdx.y;
  const int n = blockIdx.x * blockDim.x + threadIdx.x;
  const float* p = xyz1 + ((size_t)b * N_ + n) * 3;
  const float px = p[0], py = p[1], pz = p[2];

  float d0 = 3.0e38f, d1 = 3.0e38f, d2 = 3.0e38f;
  int   i0 = 0, i1 = 0, i2 = 0;

  for (int s0 = 0; s0 < S_; s0 += KTS) {
    for (int t = threadIdx.x; t < KTS; t += 256) {
      const float* q = xyz2 + ((size_t)b * S_ + s0 + t) * 3;
      sx[t] = q[0]; sy[t] = q[1]; sz[t] = q[2];
    }
    __syncthreads();
#pragma unroll 8
    for (int j = 0; j < KTS; ++j) {
      float dx = px - sx[j], dy = py - sy[j], dz = pz - sz[j];
      float d = dx*dx + dy*dy + dz*dz;
      int gi = s0 + j;
      if (d < d2) {
        if (d < d1) {
          d2 = d1; i2 = i1;
          if (d < d0) { d1 = d0; i1 = i0; d0 = d; i0 = gi; }
          else        { d1 = d;  i1 = gi; }
        } else { d2 = d; i2 = gi; }
      }
    }
    __syncthreads();
  }
  const float r0 = 1.f / (d0 + 1e-8f), r1 = 1.f / (d1 + 1e-8f), r2 = 1.f / (d2 + 1e-8f);
  const float rs = 1.f / (r0 + r1 + r2);
  const size_t o = ((size_t)b * N_ + n) * 3;
  idx[o+0] = i0; idx[o+1] = i1; idx[o+2] = i2;
  wgt[o+0] = r0 * rs; wgt[o+1] = r1 * rs; wgt[o+2] = r2 * rs;
}

// ============ kernel 3: gather + interp + concat -> bf16 X ============
__global__ void __launch_bounds__(256) gather_concat(
    const float* __restrict__ feat1, const float* __restrict__ feat2,
    const int* __restrict__ idx, const float* __restrict__ wgt,
    unsigned* __restrict__ X) {  // rows of 384 bf16 = 192 uints
  const int lane = threadIdx.x & 31;
  const int p = blockIdx.x * 8 + (threadIdx.x >> 5);
  const int b = p >> 14;  // p / N_
  unsigned* xr = X + (size_t)p * (CIN_ / 2);

  const float4 f = *(const float4*)(feat1 + (size_t)p * C1_ + lane * 4);
  xr[lane*2 + 0] = pack2bf(f.x, f.y);
  xr[lane*2 + 1] = pack2bf(f.z, f.w);

  const int   j0 = idx[p*3+0], j1 = idx[p*3+1], j2 = idx[p*3+2];
  const float w0 = wgt[p*3+0], w1 = wgt[p*3+1], w2 = wgt[p*3+2];
  const float* fb = feat2 + (size_t)b * S_ * C2_;
  const float* r0 = fb + (size_t)j0 * C2_ + lane * 8;
  const float* r1 = fb + (size_t)j1 * C2_ + lane * 8;
  const float* r2 = fb + (size_t)j2 * C2_ + lane * 8;
  float4 a0 = *(const float4*)(r0), a1 = *(const float4*)(r0 + 4);
  float4 b0 = *(const float4*)(r1), b1 = *(const float4*)(r1 + 4);
  float4 c0 = *(const float4*)(r2), c1 = *(const float4*)(r2 + 4);
  float o[8];
  o[0] = w0*a0.x + w1*b0.x + w2*c0.x;  o[1] = w0*a0.y + w1*b0.y + w2*c0.y;
  o[2] = w0*a0.z + w1*b0.z + w2*c0.z;  o[3] = w0*a0.w + w1*b0.w + w2*c0.w;
  o[4] = w0*a1.x + w1*b1.x + w2*c1.x;  o[5] = w0*a1.y + w1*b1.y + w2*c1.y;
  o[6] = w0*a1.z + w1*b1.z + w2*c1.z;  o[7] = w0*a1.w + w1*b1.w + w2*c1.w;
  unsigned* xi = xr + C1_ / 2;
#pragma unroll
  for (int j = 0; j < 4; ++j) xi[lane*4 + j] = pack2bf(o[2*j], o[2*j+1]);
}

// A-fragment load (global): lane holds row (l&15), K chunk
// [kb+16*hi, +16)
__device__ __forceinline__ v16bf load_afrag(const __bf16* xrow, int kb, int hi) {
  uint4 q0 = *(const uint4*)(xrow + kb + hi * 16);
  uint4 q1 = *(const uint4*)(xrow + kb + hi * 16 + 8);
  unsigned au[8] = {q0.x, q0.y, q0.z, q0.w, q1.x, q1.y, q1.z, q1.w};
  v16bf a; __builtin_memcpy(&a, au, 32);
  return a;
}

__device__ __forceinline__ v16bf load_bfrag_lds(const unsigned short* bp) {
  v16bf bf; __builtin_memcpy(&bf, bp, 32);  // ds_load_b128 x2
  return bf;
}

// ================= kernel 4: GEMM1 384->256 + BN + ReLU, bf16 out =========
// one wave per 16-row M tile; W k-slices double-buffered in LDS via async copy
#define RS1 40   // LDS row stride in bf16 (80B: 16B-aligned halves, bank-spread)
__global__ void __launch_bounds__(256) gemm_l1(
    const __bf16* __restrict__ X, const __bf16* __restrict__ W,
    const float* __restrict__ bb, const float* __restrict__ gg,
    const float* __restrict__ be, const float* __restrict__ mm,
    const float* __restrict__ vv, __bf16* __restrict__ Y) {
  __shared__ __align__(16) unsigned short sW[2][H1_ * RS1]; // 2 x 20KB
  const int tid  = threadIdx.x;
  const int lane = tid & 31;
  const int wv   = blockIdx.x * 8 + (tid >> 5);
  const int row  = lane & 15;
  const int hi   = lane >> 4;
  const size_t mBase = (size_t)wv * 16;

  // staging bases: thread tid owns W row tid (64B per k-slice)
  const char* gRow = (const char*)(W + (size_t)tid * CIN_);
  const unsigned laBuf0 = lds_addr_of(&sW[0][tid * RS1]);
  const unsigned laBuf1 = lds_addr_of(&sW[1][tid * RS1]);
  // B-fragment read bases (per-nt term is a constant -> ds offset immediate)
  const unsigned short* bp0 = &sW[0][row * RS1 + hi * 16];
  const unsigned short* bp1 = &sW[1][row * RS1 + hi * 16];

  v8f acc[16];
#pragma unroll
  for (int t = 0; t < 16; ++t)
#pragma unroll
    for (int r = 0; r < 8; ++r) acc[t][r] = 0.f;

  const __bf16* xrow = X + (mBase + row) * CIN_;

  async_g2l_row64(laBuf0, gRow);
  v16bf aCur = load_afrag(xrow, 0, hi);
  wait_async0();
  __syncthreads();

  // fully unrolled: buffer selects fold to constants, B-frags have short
  // live ranges, accumulators stay pinned (no pipelined reg rotation/spill)
#pragma unroll
  for (int kk = 0; kk < CIN_ / 32; ++kk) {
    if (kk + 1 < CIN_ / 32)                                 // async, overlapped
      async_g2l_row64((kk & 1) ? laBuf0 : laBuf1, gRow + (kk + 1) * 64);
    v16bf aNext = aCur;
    if (kk + 1 < CIN_ / 32) aNext = load_afrag(xrow, (kk + 1) * 32, hi);
    const unsigned short* bp = (kk & 1) ? bp1 : bp0;
#pragma unroll
    for (int nt = 0; nt < 16; ++nt) {
      v16bf bf = load_bfrag_lds(bp + nt * (16 * RS1));
      acc[nt] = __builtin_amdgcn_wmma_f32_16x16x32_bf16(
          false, aCur, false, bf, (short)0, acc[nt], false, false);
    }
    aCur = aNext;
    wait_async0();
    __syncthreads();
  }

  // fused bias+BN+ReLU epilogue, bf16 output
#pragma unroll
  for (int nt = 0; nt < 16; ++nt) {
    const int ch = nt * 16 + row;
    const float sc = gg[ch] * rsqrtf(vv[ch] + 1e-5f);
    const float sh = (bb[ch] - mm[ch]) * sc + be[ch];
#pragma unroll
    for (int r = 0; r < 8; ++r) {
      float val = fmaxf(acc[nt][r] * sc + sh, 0.f);
      size_t m = mBase + (size_t)(r + hi * 8);
      *((unsigned short*)(Y + m * H1_ + ch)) = f2bf(val);
    }
  }
}

// ================= kernel 5: GEMM2 256->128 + BN + ReLU, f32 out ==========
#define RS2 40
__global__ void __launch_bounds__(256) gemm_l2(
    const __bf16* __restrict__ X, const __bf16* __restrict__ W,
    const float* __restrict__ bb, const float* __restrict__ gg,
    const float* __restrict__ be, const float* __restrict__ mm,
    const float* __restrict__ vv, float* __restrict__ O) {
  __shared__ __align__(16) unsigned short sW[2][COUT_ * RS2]; // 2 x 10KB
  const int tid  = threadIdx.x;
  const int lane = tid & 31;
  const int wv   = blockIdx.x * 8 + (tid >> 5);
  const int row  = lane & 15;
  const int hi   = lane >> 4;
  const size_t mBase = (size_t)wv * 16;

  const char* gRow = (const char*)(W + (size_t)tid * H1_);
  const unsigned laBuf0 = lds_addr_of(&sW[0][tid * RS2]);
  const unsigned laBuf1 = lds_addr_of(&sW[1][tid * RS2]);
  const unsigned short* bp0 = &sW[0][row * RS2 + hi * 16];
  const unsigned short* bp1 = &sW[1][row * RS2 + hi * 16];

  v8f acc[8];
#pragma unroll
  for (int t = 0; t < 8; ++t)
#pragma unroll
    for (int r = 0; r < 8; ++r) acc[t][r] = 0.f;

  const __bf16* xrow = X + (mBase + row) * H1_;

  if (tid < COUT_) async_g2l_row64(laBuf0, gRow);
  v16bf aCur = load_afrag(xrow, 0, hi);
  wait_async0();
  __syncthreads();

#pragma unroll
  for (int kk = 0; kk < H1_ / 32; ++kk) {
    if (tid < COUT_ && kk + 1 < H1_ / 32)
      async_g2l_row64((kk & 1) ? laBuf0 : laBuf1, gRow + (kk + 1) * 64);
    v16bf aNext = aCur;
    if (kk + 1 < H1_ / 32) aNext = load_afrag(xrow, (kk + 1) * 32, hi);
    const unsigned short* bp = (kk & 1) ? bp1 : bp0;
#pragma unroll
    for (int nt = 0; nt < 8; ++nt) {
      v16bf bf = load_bfrag_lds(bp + nt * (16 * RS2));
      acc[nt] = __builtin_amdgcn_wmma_f32_16x16x32_bf16(
          false, aCur, false, bf, (short)0, acc[nt], false, false);
    }
    aCur = aNext;
    wait_async0();
    __syncthreads();
  }

#pragma unroll
  for (int nt = 0; nt < 8; ++nt) {
    const int ch = nt * 16 + row;
    const float sc = gg[ch] * rsqrtf(vv[ch] + 1e-5f);
    const float sh = (bb[ch] - mm[ch]) * sc + be[ch];
#pragma unroll
    for (int r = 0; r < 8; ++r) {
      float val = fmaxf(acc[nt][r] * sc + sh, 0.f);
      size_t m = mBase + (size_t)(r + hi * 8);
      O[m * COUT_ + ch] = val;
    }
  }
}

// ================= host launcher =================
extern "C" void kernel_launch(void* const* d_in, const int* in_sizes, int n_in,
                              void* d_out, int out_size, void* d_ws, size_t ws_size,
                              hipStream_t stream) {
  const float* xyz1  = (const float*)d_in[0];
  const float* feat1 = (const float*)d_in[1];
  const float* xyz2  = (const float*)d_in[2];
  const float* feat2 = (const float*)d_in[3];
  const float* W1 = (const float*)d_in[4];
  const float* b1 = (const float*)d_in[5];
  const float* g1 = (const float*)d_in[6];
  const float* be1= (const float*)d_in[7];
  const float* m1 = (const float*)d_in[8];
  const float* v1 = (const float*)d_in[9];
  const float* W2 = (const float*)d_in[10];
  const float* b2 = (const float*)d_in[11];
  const float* g2 = (const float*)d_in[12];
  const float* be2= (const float*)d_in[13];
  const float* m2 = (const float*)d_in[14];
  const float* v2 = (const float*)d_in[15];
  float* out = (float*)d_out;

  char* ws = (char*)d_ws;
  size_t off = 0;
  auto take = [&](size_t bytes) -> char* {
    char* p = ws + off;
    off += (bytes + 255) & ~(size_t)255;
    return p;
  };
  int*      d_idx = (int*)     take((size_t)R_ * 3 * 4);
  float*    d_w   = (float*)   take((size_t)R_ * 3 * 4);
  unsigned* d_X   = (unsigned*)take((size_t)R_ * CIN_ * 2);
  __bf16*   d_Y   = (__bf16*)  take((size_t)R_ * H1_ * 2);
  unsigned* d_W1b = (unsigned*)take((size_t)H1_ * CIN_ * 2);
  unsigned* d_W2b = (unsigned*)take((size_t)COUT_ * H1_ * 2);
  (void)ws_size; (void)in_sizes; (void)n_in; (void)out_size;

  cvt_weights<<<dim3((H1_ * CIN_ / 2 + 255) / 256), dim3(256), 0, stream>>>(
      W1, W2, d_W1b, d_W2b);

  knn3<<<dim3(N_ / 256, B_), dim3(256), 0, stream>>>(xyz1, xyz2, d_idx, d_w);

  gather_concat<<<dim3(R_ / 8), dim3(256), 0, stream>>>(
      feat1, feat2, d_idx, d_w, d_X);

  gemm_l1<<<dim3(R_ / 16 / 8), dim3(256), 0, stream>>>(
      (const __bf16*)d_X, (const __bf16*)d_W1b, b1, g1, be1, m1, v1, d_Y);

  gemm_l2<<<dim3(R_ / 16 / 8), dim3(256), 0, stream>>>(
      d_Y, (const __bf16*)d_W2b, b2, g2, be2, m2, v2, out);
}
